// CadenceGNNNeighbor_87033217286453
// MI455X (gfx1250) — matrix operand order repaired and checked by reading, not compiled, not run on hardware
//
#include <hip/hip_runtime.h>
#include <hip/hip_bf16.h>

typedef __bf16 bf16_t;
typedef __attribute__((ext_vector_type(16))) __bf16 v16bf;
typedef __attribute__((ext_vector_type(8)))  __bf16 v8bf;
typedef __attribute__((ext_vector_type(8)))  float  v8f;

#define USE_ASYNC_B_STAGE 1

// Async global->LDS copy of 16B (GLOBAL_LOAD_ASYNC_TO_LDS_B128, ASYNCcnt).
__device__ __forceinline__ void async_copy16B_g2l(unsigned ldsByteOff, const void* gaddr) {
  asm volatile("global_load_async_to_lds_b128 %0, %1, off"
               :: "v"(ldsByteOff), "v"(gaddr)
               : "memory");
}
__device__ __forceinline__ void wait_asynccnt0() {
  asm volatile("s_wait_asynccnt 0x0" ::: "memory");
}

// ---------------------------------------------------------------------------
// Elementwise helpers
// ---------------------------------------------------------------------------
__global__ void k_zero_f32(float* __restrict__ p, long n) {
  long i = (long)blockIdx.x * 256 + threadIdx.x;
  if (i < n) p[i] = 0.f;
}

__global__ void k_edge_count(const int* __restrict__ dst, float* __restrict__ cnt, int E) {
  int i = blockIdx.x * 256 + threadIdx.x;
  if (i < E) atomicAdd(&cnt[dst[i]], 1.f);
}

__global__ void k_invert_cnt(float* __restrict__ c, long n) {
  long i = (long)blockIdx.x * 256 + threadIdx.x;
  if (i < n) c[i] = 1.f / fmaxf(c[i], 1.f);
}

// scatter-add D=2^logD floats per edge; agg[dst] += h[src]
__global__ void k_scatter_add(const float* __restrict__ h, const int* __restrict__ src,
                              const int* __restrict__ dst, float* __restrict__ agg,
                              int E, int logD) {
  const int logDq = logD - 2;
  long i = (long)blockIdx.x * 256 + threadIdx.x;
  long total = (long)E << logDq;
  if (i >= total) return;
  int e = (int)(i >> logDq);
  int d = (int)(i & ((1 << logDq) - 1)) << 2;
  const float4 v = *(const float4*)(h + ((long)src[e] << logD) + d);
  float* a = agg + ((long)dst[e] << logD) + d;
  atomicAdd(a + 0, v.x);
  atomicAdd(a + 1, v.y);
  atomicAdd(a + 2, v.z);
  atomicAdd(a + 3, v.w);
}

// dst[n*ldd + colofs + d] = bf16(src[n*D + d] * (inv ? inv[n] : 1)), D = 2^logD
__global__ void k_scale_cvt(const float* __restrict__ src, const float* __restrict__ inv,
                            bf16_t* __restrict__ dstp, int ldd, int colofs,
                            int N, int logD) {
  long i = (long)blockIdx.x * 256 + threadIdx.x;
  long total = (long)N << logD;
  if (i >= total) return;
  int n = (int)(i >> logD);
  int d = (int)(i & ((1 << logD) - 1));
  float sc = inv ? inv[n] : 1.f;
  dstp[(long)n * ldd + colofs + d] = (bf16_t)(src[i] * sc);
}

// ---------------------------------------------------------------------------
// Weight packing into per-lane WMMA B-fragment layout.
// Bp[(kt*Nt + nt)*32 + lane][e] (16 bf16 = 32B contiguous per lane), where
// element e of lane L is B[kt*32 + (e&8)*2 + (L>>4)*8 + (e&7)][nt*16 + (L&15)].
// ---------------------------------------------------------------------------
__global__ void k_pack_bfrag(const float* __restrict__ W, bf16_t* __restrict__ Bp,
                             int K, int Nc, int logNt) {
  long i = (long)blockIdx.x * 256 + threadIdx.x;
  long total = (long)K * Nc;
  if (i >= total) return;
  int  e    = (int)(i & 15);
  long r    = i >> 4;
  int  lane = (int)(r & 31);
  r >>= 5;
  int nt = (int)(r & ((1 << logNt) - 1));
  int kt = (int)(r >> logNt);
  int k  = kt * 32 + (e & 8) * 2 + (lane >> 4) * 8 + (e & 7);
  int c  = nt * 16 + (lane & 15);
  Bp[i] = (bf16_t)W[(long)k * Nc + c];
}

// Same, from the virtual concat [Wl_0;..;Wl_3;sum_t Wr_t]; K = 2^logK.
__global__ void k_pack_wcat_frag(const float* __restrict__ Wl, const float* __restrict__ Wr,
                                 bf16_t* __restrict__ Bp, int logK, int H, int logNt, int T) {
  const int K = 1 << logK;
  long i = (long)blockIdx.x * 256 + threadIdx.x;
  long total = (long)(T + 1) * K * H;
  if (i >= total) return;
  int  e    = (int)(i & 15);
  long r    = i >> 4;
  int  lane = (int)(r & 31);
  r >>= 5;
  int nt   = (int)(r & ((1 << logNt) - 1));
  int kt   = (int)(r >> logNt);
  int krow = kt * 32 + (e & 8) * 2 + (lane >> 4) * 8 + (e & 7);
  int c    = nt * 16 + (lane & 15);
  int blk  = krow >> logK;
  int k    = krow & (K - 1);
  float v;
  if (blk < T) {
    v = Wl[((long)blk * K + k) * H + c];
  } else {
    v = 0.f;
    for (int t = 0; t < T; ++t) v += Wr[((long)t * K + k) * H + c];
  }
  Bp[i] = (bf16_t)v;
}

__global__ void k_sum_bias(const float* __restrict__ bl, float* __restrict__ bs, int H, int T) {
  int i = blockIdx.x * 256 + threadIdx.x;
  if (i >= H) return;
  float v = 0.f;
  for (int t = 0; t < T; ++t) v += bl[t * H + i];
  bs[i] = v;
}

// out = (g + s) * inv[n]   (onset pooling), D = 2^logD
__global__ void k_pool(const float* __restrict__ g, const float* __restrict__ s,
                       const float* __restrict__ inv, float* __restrict__ out,
                       long total, int logD) {
  long i = (long)blockIdx.x * 256 + threadIdx.x;
  if (i >= total) return;
  int n = (int)(i >> logD);
  out[i] = (g[i] + s[i]) * inv[n];
}

// ---------------------------------------------------------------------------
// WMMA GEMM:  C[M,Nc] = act(A[M,K] @ B[K,Nc] + bias)
// A: bf16 row-major (streamed, prefetched). B: pre-swizzled fragments, staged
// per 4-K-step chunk into LDS with GLOBAL_LOAD_ASYNC_TO_LDS_B128 (all 8 waves
// in a block share the same B panel). One wave32 computes a 16x32 tile: two
// f32 accumulators per shared A fragment.
// ---------------------------------------------------------------------------
#define KCHUNK 4   // K-steps per LDS stage; KCHUNK*64 units == 256 threads
static_assert(KCHUNK * 64 == 256, "one 32B unit per thread");

__global__ __launch_bounds__(256)
void k_wmma_gemm(const bf16_t* __restrict__ A, int lda,
                 const bf16_t* __restrict__ Bpacked,
                 const float* __restrict__ bias,
                 float* __restrict__ C, int ldc,
                 int M, int K, int Nc, int relu) {
  __shared__ __align__(32) bf16_t sB[KCHUNK * 2 * 32 * 16];   // 8KB

  const int tid  = threadIdx.y * 32 + threadIdx.x;
  const int lane = threadIdx.x;       // 0..31
  const int l16  = lane & 15;
  const int hi   = lane >> 4;
  const int tileM  = (blockIdx.x * blockDim.y + threadIdx.y) * 16;
  const int ntPair = blockIdx.y;      // 32-wide column pair (shared by block)
  const int Nt     = Nc >> 4;
  const int kSteps = K >> 5;          // multiple of KCHUNK for all shapes here
  const bool rowOK = (tileM < M);     // wave-uniform

  v8f acc0 = {}, acc1 = {};

  // A fragment (16-bit A 16x32): lane l16 = row; element K offsets are
  // [8*hi, 8*hi+8) and [16+8*hi, 16+8*hi+8) -> two 16B runs.
  const bf16_t* aptr = A + (size_t)(rowOK ? (tileM + l16) : l16) * lda + hi * 8;

  // Cooperative B staging: unit u (32B) = [ktl][frag][lane] of current chunk.
  const int u    = tid;
  const int ktl  = u >> 6;
  const int frag = (u >> 5) & 1;
  const int ln   = u & 31;
  const bf16_t* bsrc = Bpacked + ((size_t)(ktl * Nt + 2 * ntPair + frag) * 32 + ln) * 16;
  const size_t bsrcStep = (size_t)KCHUNK * Nt * 32 * 16;   // elems per chunk
  const unsigned ldsOff = (unsigned)(size_t)&sB[u * 16];

  for (int kc = 0; kc < kSteps; kc += KCHUNK) {
    __syncthreads();                   // previous chunk fully consumed
#if USE_ASYNC_B_STAGE
    async_copy16B_g2l(ldsOff,      bsrc);
    async_copy16B_g2l(ldsOff + 16, bsrc + 8);
    wait_asynccnt0();
#else
    *(v16bf*)&sB[u * 16] = *(const v16bf*)bsrc;
#endif
    bsrc += bsrcStep;
    __syncthreads();                   // chunk visible to all waves

    if (rowOK) {
#pragma unroll
      for (int k = 0; k < KCHUNK; ++k) {
        union { v16bf v; v8bf h[2]; } a;
        a.h[0] = *(const v8bf*)(aptr);
        a.h[1] = *(const v8bf*)(aptr + 16);
        __builtin_prefetch(aptr + 32 * KCHUNK, 0, 1);   // next chunk of A
        v16bf bf0 = ((const v16bf*)sB)[(k * 2 + 0) * 32 + lane];
        v16bf bf1 = ((const v16bf*)sB)[(k * 2 + 1) * 32 + lane];
        acc0 = __builtin_amdgcn_wmma_f32_16x16x32_bf16(
            false, a.v, false, bf0, (short)0, acc0, false, false);
        acc1 = __builtin_amdgcn_wmma_f32_16x16x32_bf16(
            false, a.v, false, bf1, (short)0, acc1, false, false);
        aptr += 32;
      }
    }
  }

  if (!rowOK) return;

  // C/D f32 16x16: lane -> col = lane%16; VGPR v -> row = v + 8*hi
  const int col0 = ntPair * 32 + l16;
  const int col1 = col0 + 16;
  const float bv0 = bias ? bias[col0] : 0.f;
  const float bv1 = bias ? bias[col1] : 0.f;
#pragma unroll
  for (int v = 0; v < 8; ++v) {
    int row = tileM + v + hi * 8;
    float x0 = acc0[v] + bv0;
    float x1 = acc1[v] + bv1;
    if (relu) { x0 = fmaxf(x0, 0.f); x1 = fmaxf(x1, 0.f); }
    C[(size_t)row * ldc + col0] = x0;
    C[(size_t)row * ldc + col1] = x1;
  }
}

// ---------------------------------------------------------------------------
// LayerNorm over D=128, one wave32 per row (4 floats/lane), shuffle reduce.
// ---------------------------------------------------------------------------
__global__ void k_layernorm128(float* __restrict__ h, const float* __restrict__ g,
                               const float* __restrict__ b, bf16_t* __restrict__ obf,
                               int N) {
  const int lane = threadIdx.x;
  const int row  = blockIdx.x * blockDim.y + threadIdx.y;
  if (row >= N) return;
  float4 v = ((const float4*)(h + (long)row * 128))[lane];
  float s = v.x + v.y + v.z + v.w;
  float q = v.x * v.x + v.y * v.y + v.z * v.z + v.w * v.w;
#pragma unroll
  for (int o = 16; o >= 1; o >>= 1) {
    s += __shfl_xor(s, o, 32);
    q += __shfl_xor(q, o, 32);
  }
  const float mu  = s * (1.f / 128.f);
  const float var = q * (1.f / 128.f) - mu * mu;
  const float r   = rsqrtf(var + 1e-5f);
  float4 gg = ((const float4*)g)[lane];
  float4 bb = ((const float4*)b)[lane];
  float4 o;
  o.x = (v.x - mu) * r * gg.x + bb.x;
  o.y = (v.y - mu) * r * gg.y + bb.y;
  o.z = (v.z - mu) * r * gg.z + bb.z;
  o.w = (v.w - mu) * r * gg.w + bb.w;
  ((float4*)(h + (long)row * 128))[lane] = o;
  if (obf) {
    bf16_t* op = obf + (long)row * 128 + lane * 4;
    op[0] = (bf16_t)o.x; op[1] = (bf16_t)o.y;
    op[2] = (bf16_t)o.z; op[3] = (bf16_t)o.w;
  }
}

// ---------------------------------------------------------------------------
// BatchNorm batch statistics: one block per channel, LDS tree reduce.
// ---------------------------------------------------------------------------
__global__ void k_bn_stats(const float* __restrict__ z, float* __restrict__ stats,
                           int N, int C) {
  const int ch = blockIdx.x;
  float s = 0.f, q = 0.f;
  for (int n = threadIdx.x; n < N; n += blockDim.x) {
    float v = z[(long)n * C + ch];
    s += v;
    q += v * v;
  }
  __shared__ float sh[256];
  __shared__ float sq[256];
  sh[threadIdx.x] = s;
  sq[threadIdx.x] = q;
  __syncthreads();
  for (int st = 128; st > 0; st >>= 1) {
    if ((int)threadIdx.x < st) {
      sh[threadIdx.x] += sh[threadIdx.x + st];
      sq[threadIdx.x] += sq[threadIdx.x + st];
    }
    __syncthreads();
  }
  if (threadIdx.x == 0) {
    float mu  = sh[0] / (float)N;
    float var = sq[0] / (float)N - mu * mu;
    stats[ch]     = mu;
    stats[C + ch] = rsqrtf(var + 1e-5f);
  }
}

// ---------------------------------------------------------------------------
// Final: BN-normalize z[N,64], 64x7 matvec, softmax. One thread per node.
// ---------------------------------------------------------------------------
__global__ void k_clf_final(const float* __restrict__ z, const float* __restrict__ stats,
                            const float* __restrict__ g, const float* __restrict__ b,
                            const float* __restrict__ W2, const float* __restrict__ b2,
                            float* __restrict__ out, int N) {
  int n = blockIdx.x * 256 + threadIdx.x;
  if (n >= N) return;
  float acc[7];
#pragma unroll
  for (int o = 0; o < 7; ++o) acc[o] = b2[o];
  for (int i = 0; i < 64; ++i) {
    float v = (z[(long)n * 64 + i] - stats[i]) * stats[64 + i] * g[i] + b[i];
#pragma unroll
    for (int o = 0; o < 7; ++o) acc[o] += v * W2[i * 7 + o];
  }
  float m = acc[0];
#pragma unroll
  for (int o = 1; o < 7; ++o) m = fmaxf(m, acc[o]);
  float ssum = 0.f;
#pragma unroll
  for (int o = 0; o < 7; ++o) { acc[o] = __expf(acc[o] - m); ssum += acc[o]; }
  float r = 1.f / ssum;
#pragma unroll
  for (int o = 0; o < 7; ++o) out[(long)n * 7 + o] = acc[o] * r;
}

// ---------------------------------------------------------------------------
// Launch
// ---------------------------------------------------------------------------
extern "C" void kernel_launch(void* const* d_in, const int* in_sizes, int n_in,
                              void* d_out, int out_size, void* d_ws, size_t ws_size,
                              hipStream_t stream) {
  const int IN = 128, HID = 256, OH = 128, CH = 64, T = 4;
  const int N = in_sizes[0] / IN;
  const int E = in_sizes[1] / 2;

  const float* x      = (const float*)d_in[0];
  const int*   edges[4] = {(const int*)d_in[1], (const int*)d_in[2],
                           (const int*)d_in[3], (const int*)d_in[4]};
  const float* Wl0  = (const float*)d_in[5];
  const float* bl0  = (const float*)d_in[6];
  const float* Wr0  = (const float*)d_in[7];
  const float* Wl   = (const float*)d_in[8];
  const float* bl   = (const float*)d_in[9];
  const float* Wr   = (const float*)d_in[10];
  const float* linW = (const float*)d_in[11];
  const float* linb = (const float*)d_in[12];
  const float* ng   = (const float*)d_in[13];
  const float* nb   = (const float*)d_in[14];
  const float* pW1  = (const float*)d_in[15];
  const float* pb1  = (const float*)d_in[16];
  const float* plng = (const float*)d_in[17];
  const float* plnb = (const float*)d_in[18];
  const float* pW2  = (const float*)d_in[19];
  const float* pb2  = (const float*)d_in[20];
  const float* cW1  = (const float*)d_in[21];
  const float* cb1  = (const float*)d_in[22];
  const float* bng  = (const float*)d_in[23];
  const float* bnb  = (const float*)d_in[24];
  const float* cW2  = (const float*)d_in[25];
  const float* cb2  = (const float*)d_in[26];
  float* out = (float*)d_out;

  // workspace carve
  char* wp = (char*)d_ws;
  auto carve = [&](size_t bytes) -> char* {
    char* p = wp;
    wp += (bytes + 255) & ~(size_t)255;
    return p;
  };
  float*  inv   = (float*)carve((size_t)4 * N * 4);
  bf16_t* Xcat  = (bf16_t*)carve((size_t)N * 1280 * 2);
  float*  agg   = (float*)carve((size_t)N * 256 * 4);
  float*  h     = (float*)carve((size_t)N * 256 * 4);
  float*  g     = (float*)carve((size_t)N * 128 * 4);
  float*  sbuf  = (float*)carve((size_t)N * 128 * 4);
  float*  t1    = (float*)carve((size_t)N * 128 * 4);
  bf16_t* hb    = (bf16_t*)carve((size_t)N * 128 * 2);
  float*  z     = (float*)carve((size_t)N * 64 * 4);
  float*  bnst  = (float*)carve(2 * 64 * 4);
  bf16_t* Wc0   = (bf16_t*)carve((size_t)5 * IN * HID * 2);
  bf16_t* Wc1   = (bf16_t*)carve((size_t)5 * HID * HID * 2);
  bf16_t* Wc2   = (bf16_t*)carve((size_t)5 * HID * HID * 2);
  float*  bs0   = (float*)carve(HID * 4);
  float*  bs1   = (float*)carve(HID * 4);
  float*  bs2   = (float*)carve(HID * 4);
  bf16_t* linWb = (bf16_t*)carve((size_t)HID * OH * 2);
  bf16_t* pW1b  = (bf16_t*)carve((size_t)OH * OH * 2);
  bf16_t* pW2b  = (bf16_t*)carve((size_t)OH * OH * 2);
  bf16_t* cW1b  = (bf16_t*)carve((size_t)OH * CH * 2);

  auto g1 = [](long n) { return dim3((unsigned)((n + 255) / 256)); };
  const dim3 gblk(32, 8);   // 8 waves -> 8 row tiles per block
  const dim3 lnblk(32, 8);

  // ---- weight prep: convert + swizzle into WMMA B-fragment layout ----
  // logNt: HID->16 cols/16 = 16 -> 4 ; OH -> 8 -> 3 ; CH -> 4 -> 2
  k_pack_wcat_frag<<<g1((long)5 * IN * HID), 256, 0, stream>>>(Wl0, Wr0, Wc0, 7, HID, 4, T);
  k_sum_bias<<<g1(HID), 256, 0, stream>>>(bl0, bs0, HID, T);
  k_pack_wcat_frag<<<g1((long)5 * HID * HID), 256, 0, stream>>>(Wl, Wr, Wc1, 8, HID, 4, T);
  k_sum_bias<<<g1(HID), 256, 0, stream>>>(bl, bs1, HID, T);
  k_pack_wcat_frag<<<g1((long)5 * HID * HID), 256, 0, stream>>>(
      Wl + (size_t)T * HID * HID, Wr + (size_t)T * HID * HID, Wc2, 8, HID, 4, T);
  k_sum_bias<<<g1(HID), 256, 0, stream>>>(bl + T * HID, bs2, HID, T);
  k_pack_bfrag<<<g1((long)HID * OH), 256, 0, stream>>>(linW, linWb, HID, OH, 3);
  k_pack_bfrag<<<g1((long)OH * OH), 256, 0, stream>>>(pW1, pW1b, OH, OH, 3);
  k_pack_bfrag<<<g1((long)OH * OH), 256, 0, stream>>>(pW2, pW2b, OH, OH, 3);
  k_pack_bfrag<<<g1((long)OH * CH), 256, 0, stream>>>(cW1, cW1b, OH, CH, 2);

  // ---- degree counts -> reciprocals ----
  k_zero_f32<<<g1((long)4 * N), 256, 0, stream>>>(inv, (long)4 * N);
  for (int t = 0; t < 4; ++t)
    k_edge_count<<<g1(E), 256, 0, stream>>>(edges[t] + E, inv + (long)t * N, E);
  k_invert_cnt<<<g1((long)4 * N), 256, 0, stream>>>(inv, (long)4 * N);

  // ---- layer 0 (input dim 128 = 2^7, Xcat ld = 640) ----
  for (int t = 0; t < 4; ++t) {
    k_zero_f32<<<g1((long)N * IN), 256, 0, stream>>>(agg, (long)N * IN);
    k_scatter_add<<<g1((long)E * (IN / 4)), 256, 0, stream>>>(x, edges[t], edges[t] + E, agg, E, 7);
    k_scale_cvt<<<g1((long)N * IN), 256, 0, stream>>>(agg, inv + (long)t * N, Xcat, 5 * IN, t * IN, N, 7);
  }
  k_scale_cvt<<<g1((long)N * IN), 256, 0, stream>>>(x, nullptr, Xcat, 5 * IN, 4 * IN, N, 7);
  k_wmma_gemm<<<dim3((N / 16 + 7) / 8, HID / 32), gblk, 0, stream>>>(
      Xcat, 5 * IN, Wc0, bs0, h, HID, N, 5 * IN, HID, 1);

  // ---- layers 1..2 (dim 256 = 2^8, Xcat ld = 1280) ----
  bf16_t* Wcs[2] = {Wc1, Wc2};
  float*  bss[2] = {bs1, bs2};
  for (int l = 0; l < 2; ++l) {
    for (int t = 0; t < 4; ++t) {
      k_zero_f32<<<g1((long)N * HID), 256, 0, stream>>>(agg, (long)N * HID);
      k_scatter_add<<<g1((long)E * (HID / 4)), 256, 0, stream>>>(h, edges[t], edges[t] + E, agg, E, 8);
      k_scale_cvt<<<g1((long)N * HID), 256, 0, stream>>>(agg, inv + (long)t * N, Xcat, 5 * HID, t * HID, N, 8);
    }
    k_scale_cvt<<<g1((long)N * HID), 256, 0, stream>>>(h, nullptr, Xcat, 5 * HID, 4 * HID, N, 8);
    k_wmma_gemm<<<dim3((N / 16 + 7) / 8, HID / 32), gblk, 0, stream>>>(
        Xcat, 5 * HID, Wcs[l], bss[l], h, HID, N, 5 * HID, HID, 1);
  }

  // ---- gnn.lin: h[N,256] @ linW[256,128] + b ----
  k_scale_cvt<<<g1((long)N * HID), 256, 0, stream>>>(h, nullptr, Xcat, HID, 0, N, 8);
  k_wmma_gemm<<<dim3((N / 16 + 7) / 8, OH / 32), gblk, 0, stream>>>(
      Xcat, HID, linWb, linb, g, OH, N, HID, OH, 0);

  // ---- onset pooling: (g + scatter_sum(g)) / max(cnt_onset, 1) ----
  k_zero_f32<<<g1((long)N * OH), 256, 0, stream>>>(sbuf, (long)N * OH);
  k_scatter_add<<<g1((long)E * (OH / 4)), 256, 0, stream>>>(g, edges[0], edges[0] + E, sbuf, E, 7);
  k_pool<<<g1((long)N * OH), 256, 0, stream>>>(g, sbuf, inv, t1, (long)N * OH, 7);

  // ---- norm + pool_mlp ----
  k_layernorm128<<<dim3((N + 7) / 8), lnblk, 0, stream>>>(t1, ng, nb, hb, N);
  k_wmma_gemm<<<dim3((N / 16 + 7) / 8, OH / 32), gblk, 0, stream>>>(
      hb, OH, pW1b, pb1, t1, OH, N, OH, OH, 1);
  k_layernorm128<<<dim3((N + 7) / 8), lnblk, 0, stream>>>(t1, plng, plnb, hb, N);
  k_wmma_gemm<<<dim3((N / 16 + 7) / 8, OH / 32), gblk, 0, stream>>>(
      hb, OH, pW2b, pb2, t1, OH, N, OH, OH, 0);

  // ---- cad_clf: Linear(128->64) + ReLU, BN, Linear(64->7), softmax ----
  k_scale_cvt<<<g1((long)N * OH), 256, 0, stream>>>(t1, nullptr, hb, OH, 0, N, 7);
  k_wmma_gemm<<<dim3((N / 16 + 7) / 8, CH / 32), gblk, 0, stream>>>(
      hb, OH, cW1b, cb1, z, CH, N, OH, CH, 1);
  k_bn_stats<<<dim3(CH), dim3(256), 0, stream>>>(z, bnst, N, CH);
  k_clf_final<<<g1(N), 256, 0, stream>>>(z, bnst, bng, bnb, cW2, cb2, out, N);
}
